// ParallelGripperPoseSampler2_69458211111054
// MI455X (gfx1250) — matrix operand order repaired and checked by reading, not compiled, not run on hardware
//
#include <hip/hip_runtime.h>

typedef __attribute__((ext_vector_type(16))) _Float16 v16h;
typedef __attribute__((ext_vector_type(8)))  float    v8f;
typedef __attribute__((ext_vector_type(4)))  unsigned int u32x4;

#define DEVFN __device__ __forceinline__

// ---------------- problem geometry ----------------
constexpr int BHW  = 224*224;            // 50176
constexpr int NPIX = 8*BHW;              // 401408
constexpr int WAVES_PER_BLOCK = 8;       // 256 threads, wave32
constexpr int PIX_PER_BLOCK   = WAVES_PER_BLOCK*16;   // 128
constexpr int GRID            = NPIX / PIX_PER_BLOCK; // 3136, exact

// ---------------- packed weights in d_ws (f16 halves region) ----------------
constexpr int AW1   = 0;                 // appr W1 [32][64]
constexpr int AW2   = AW1 + 32*64;       // appr W2 [16][32]
constexpr int AW3   = AW2 + 16*32;       // appr W3 [16][32] (3 rows live, K 16..31 zero)
constexpr int FILM0 = AW3 + 16*32;       // 3072
constexpr int FW1o = 0;                  // film W1 [32][64]
constexpr int FWFo = 2048;               // film Wf [64][96] (columns scattered to unified enc layout)
constexpr int FW2o = 8192;               // film W2 [16][32]
constexpr int FW3o = 8704;               // film W3 [16][32]
constexpr int FSTRIDE = 9216;
constexpr int WS_HALVES = FILM0 + 3*FSTRIDE;          // 30720 halves = 61440 B

// ---------------- float region (biases/scalars), after halves ----------------
constexpr int F_AB1 = 0, F_AB2 = 32, F_AB3 = 48;      // appr b1[32], b2[16], b3(pad16)
constexpr int F_FILM = 64, F_FSTR = 136;              // per film block
constexpr int FB1o=0, FB2o=32, FB3o=48, FBFo=64, FW01o=128, FW02o=129;
constexpr int F_CENT = F_FILM + 3*F_FSTR;             // 472: centers (pad 16)
constexpr int F_MISC = F_CENT + 16;                   // 488: scale, sigma, a_w0_1, a_w0_2
constexpr int WS_FLOATS = 512;
// total ws usage: 61440 + 2048 = 63488 bytes

// unified encoding channel layout (96, zero padded):
// enc_a 0..26 | enc_b 27..44 | enc_w 45..54 | enc_d 55..64 | zeros 65..95
// film 0 = beta (in 37), film 1 = width (in 55), film 2 = dist (in 65)

struct PackArgs {
  const float *aW1,*aW2,*aW3,*aB1,*aB2,*aB3,*aw01,*aw02;
  const float *fW1[3],*fW2[3],*fW3[3],*fWf[3];
  const float *fB1[3],*fB2[3],*fB3[3],*fBf[3],*fw01[3],*fw02[3];
  const float *centers,*scale,*sigma;
};

DEVFN int wfColMap(int f, int col){
  if (f==0){ // beta: concat [enc_a, enc_d]
    if (col<27) return col;
    if (col>=55 && col<65) return col-28;
    return -1;
  } else if (f==1){ // width: concat [enc_a, enc_b, enc_d]
    if (col<45) return col;
    if (col>=55 && col<65) return col-10;
    return -1;
  }
  return (col<65) ? col : -1; // dist: identity [enc_a, enc_b, enc_w, enc_d]
}

__global__ void pack_kernel(PackArgs P, _Float16* wsH, float* wsF){
  const int r = blockIdx.y;
  const int i = blockIdx.x*blockDim.x + threadIdx.x;
  const int inC[3]={37,55,65};
  const int outC[3]={2,1,1};
  if (r==0){ if(i<2048) wsH[AW1+i]=(_Float16)P.aW1[i]; }
  else if (r==1){ if(i<512) wsH[AW2+i]=(_Float16)P.aW2[i]; }
  else if (r==2){ if(i<512){int o=i>>5,c=i&31;
      float v=(o<3&&c<16)?P.aW3[o*16+c]:0.f; wsH[AW3+i]=(_Float16)v;} }
  else if (r<15){
    int f=(r-3)>>2, k=(r-3)&3;
    _Float16* base = wsH + FILM0 + f*FSTRIDE;
    if (k==0){ if(i<2048) base[FW1o+i]=(_Float16)P.fW1[f][i]; }
    else if (k==1){ if(i<6144){int o=i/96,c=i-o*96; int sc=wfColMap(f,c);
        float v=(sc>=0)?P.fWf[f][o*inC[f]+sc]:0.f; base[FWFo+i]=(_Float16)v;} }
    else if (k==2){ if(i<512) base[FW2o+i]=(_Float16)P.fW2[f][i]; }
    else { if(i<512){int o=i>>5,c=i&31;
        float v=(o<outC[f]&&c<16)?P.fW3[f][o*16+c]:0.f; base[FW3o+i]=(_Float16)v;} }
  } else { // floats region
    if (i<WS_FLOATS){
      float v=0.f;
      if (i<32) v=P.aB1[i];
      else if (i<48) v=P.aB2[i-32];
      else if (i<64) v=((i-48)<3)?P.aB3[i-48]:0.f;
      else if (i<F_CENT){ int j=i-64, f=j/F_FSTR, k=j-f*F_FSTR;
        if (k<32) v=P.fB1[f][k];
        else if (k<48) v=P.fB2[f][k-32];
        else if (k<64) v=((k-48)<outC[f])?P.fB3[f][k-48]:0.f;
        else if (k<128) v=P.fBf[f][k-64];
        else if (k==128) v=*P.fw01[f];
        else if (k==129) v=*P.fw02[f];
      }
      else if (i<F_MISC) v=((i-F_CENT)<10)?P.centers[i-F_CENT]:0.f;
      else if (i==F_MISC)   v=*P.scale;
      else if (i==F_MISC+1) v=*P.sigma;
      else if (i==F_MISC+2) v=*P.aw01;
      else if (i==F_MISC+3) v=*P.aw02;
      wsF[i]=v;
    }
  }
}

// ---------------- WMMA fragment helpers (A/B: contiguous K per lane half) ----------------
// A (16x32 f16): lanes 0-15 row M=lane, halves 0..15 = K kb..kb+15
//                lanes 16-31: +16 K offset. Two ds_load_b128 per lane.
DEVFN v16h ldA(const _Float16* tile, int pitch, int kBase){
  const int l=threadIdx.x&31;
  const _Float16* p = tile + (l&15)*pitch + kBase + (l>>4)*16;
  v16h a;
  ((u32x4*)&a)[0] = *(const u32x4*)p;
  ((u32x4*)&a)[1] = *(const u32x4*)(p+8);
  return a;
}
// B (32x16 f16): lanes 0-15 col N=lane, halves = K kb..kb+15; lanes 16-31: K kb+16..kb+31.
// Reads W[out][in] rows => B = W^T exactly as WMMA expects.
DEVFN v16h ldB(const _Float16* w, int pitch, int nBase, int kBase){
  const int l=threadIdx.x&31;
  const _Float16* p = w + (nBase+(l&15))*pitch + kBase + (l>>4)*16;
  v16h b;
  ((u32x4*)&b)[0] = ((const u32x4*)p)[0];
  ((u32x4*)&b)[1] = ((const u32x4*)p)[1];
  return b;
}
#define WMMA_F16(A,B,C) __builtin_amdgcn_wmma_f32_16x16x32_f16(false,(A),false,(B),(short)0,(C),false,false)

DEVFN float actf(int mode, float w0, float v){
  if (mode==0) return __sinf(w0*v);       // ParameterizedSine
  if (mode==1) return v>=0.f ? v : w0*v;  // LeakyReLU(w0 = slope)
  return v;
}
// C frag (16x16 f32): lane ch = l&15 (+nBase); VGPR r -> pixel r + 8*(l>>4)
DEVFN void storeC(v8f c, _Float16* tile, int pitch, int nBase,
                  const float* bias, int mode, float w0){
  const int l=threadIdx.x&31, ch=l&15, hi=l>>4;
  const float b=bias[nBase+ch];
  #pragma unroll
  for(int r=0;r<8;r++){
    float v=actf(mode,w0,c[r]+b);
    tile[(r+8*hi)*pitch + nBase+ch] = (_Float16)v;
  }
}

// ---------------- one FiLM fusion block (16 pixels / wave) ----------------
DEVFN v8f film(const _Float16* wH, const float* sF, int f,
               const _Float16* feT, const _Float16* enT, _Float16* hT, float* scr,
               bool normX1, bool useSin, float slope){
  const _Float16* W1 = wH + FILM0 + f*FSTRIDE + FW1o;
  const _Float16* Wf = wH + FILM0 + f*FSTRIDE + FWFo;
  const _Float16* W2 = wH + FILM0 + f*FSTRIDE + FW2o;
  const _Float16* W3 = wH + FILM0 + f*FSTRIDE + FW3o;
  const float* FB = sF + F_FILM + f*F_FSTR;
  const int l=threadIdx.x&31, ch=l&15, hi=l>>4;

  _Float16 rn=(_Float16)1.0f;
  if (normX1){ // per-pixel L2 norm over 64 channels, folded into A as a row scale
    if (l<16){
      float ss=0.f;
      for(int c=0;c<64;c++){ float v=(float)feT[l*72+c]; ss+=v*v; }
      scr[96+l] = 1.0f/fmaxf(sqrtf(ss),1e-12f);
    }
    rn = (_Float16)scr[96+ch];
  }
  v8f h0={},h1={};
  #pragma unroll
  for(int kk=0;kk<64;kk+=32){
    v16h a = ldA(feT,72,kk);
    if (normX1){
      #pragma unroll
      for(int i=0;i<16;i++) a[i]*=rn;
    }
    h0 = WMMA_F16(a, ldB(W1,64,0 ,kk), h0);
    h1 = WMMA_F16(a, ldB(W1,64,16,kk), h1);
  }
  v8f g0={},g1={},g2={},g3={};
  #pragma unroll
  for(int kk=0;kk<96;kk+=32){
    v16h a = ldA(enT,104,kk);
    g0=WMMA_F16(a, ldB(Wf,96, 0,kk), g0);
    g1=WMMA_F16(a, ldB(Wf,96,16,kk), g1);
    g2=WMMA_F16(a, ldB(Wf,96,32,kk), g2);
    g3=WMMA_F16(a, ldB(Wf,96,48,kk), g3);
  }
  // FiLM directly on f32 accumulators: gamma=gb[0:32], betaAdd=gb[32:64]
  const float b1a=FB[FB1o+ch], b1b=FB[FB1o+16+ch];
  const float bg0=FB[FBFo+ch], bg1=FB[FBFo+16+ch];
  const float bb0=FB[FBFo+32+ch], bb1=FB[FBFo+48+ch];
  const int   mode = useSin?0:1;
  const float wa   = useSin?FB[FW01o]:slope;
  #pragma unroll
  for(int r=0;r<8;r++){
    float v0=(g0[r]+bg0)*(h0[r]+b1a)+(g2[r]+bb0);
    float v1=(g1[r]+bg1)*(h1[r]+b1b)+(g3[r]+bb1);
    v0=actf(mode,wa,v0); v1=actf(mode,wa,v1);
    const int row=r+8*hi;
    hT[row*40+ch]   =(_Float16)v0;
    hT[row*40+16+ch]=(_Float16)v1;
  }
  v8f c2={};
  c2=WMMA_F16(ldA(hT,40,0), ldB(W2,32,0,0), c2);
  storeC(c2,hT,40,0, FB+FB2o, mode, useSin?FB[FW02o]:slope); // cols 16..31 stale h * zero W3 cols = 0
  v8f c3={};
  c3=WMMA_F16(ldA(hT,40,0), ldB(W3,32,0,0), c3);
  const float b3=FB[FB3o+ch];
  #pragma unroll
  for(int r=0;r<8;r++) c3[r]+=b3;
  return c3;
}

// ---------------- main kernel ----------------
__global__ __launch_bounds__(256,1) void gripper_main(
    const float* __restrict__ feat, const float* __restrict__ depth,
    const _Float16* __restrict__ wH, const float* __restrict__ wFg,
    float* __restrict__ out)
{
  __shared__ float sF[WS_FLOATS];
  // per wave: feat 16x72 (1152h) | enc 16x104 (1664h) | h 16x40 (640h) | scr 128 f32 (256h)
  __shared__ __align__(16) _Float16 sA[WAVES_PER_BLOCK*3712];

  for (int i=threadIdx.x;i<WS_FLOATS;i+=blockDim.x) sF[i]=wFg[i];
  __syncthreads();

  const int wv=threadIdx.x>>5, l=threadIdx.x&31, ch=l&15, hi=l>>4;
  _Float16* feT = sA + wv*3712;
  _Float16* enT = feT + 1152;
  _Float16* hT  = enT + 1664;
  float*    scr = (float*)(hT + 640);

  const int p0  = blockIdx.x*PIX_PER_BLOCK + wv*16;
  const int b   = p0 / BHW;
  const int hw0 = p0 - b*BHW;                 // 16 consecutive pixels, same image
  const float* fb = feat + ((size_t)b*64)*BHW + hw0;

  // features tile: coalesced channel-major reads -> f16 [pix][ch] LDS
  for(int c=hi;c<64;c+=2)
    feT[ch*72 + c] = (_Float16)fb[(size_t)c*BHW + ch];

  // zero encoding tile (padding channels must be 0)
  for(int i=l;i<416;i+=32) ((unsigned long long*)enT)[i]=0ull;

  // depth -> enc_d (RBF), channels 55..64
  {
    const float sg=sF[F_MISC+1]; const float i2s=0.5f/(sg*sg);
    if (l<16){
      const float d = depth[(size_t)b*BHW + hw0 + l];
      #pragma unroll
      for(int c=0;c<10;c++){ float df=d-sF[F_CENT+c];
        enT[l*104+55+c]=(_Float16)__expf(-df*df*i2s); }
    }
  }

  // ---- appr MLP: 64 ->32(sin) ->16(sin) ->3, normalize+blend+normalize, enc_a ----
  {
    v8f h0={},h1={};
    #pragma unroll
    for(int kk=0;kk<64;kk+=32){
      v16h a=ldA(feT,72,kk);
      h0=WMMA_F16(a, ldB(wH+AW1,64,0 ,kk), h0);
      h1=WMMA_F16(a, ldB(wH+AW1,64,16,kk), h1);
    }
    const float w01=sF[F_MISC+2], w02=sF[F_MISC+3];
    storeC(h0,hT,40,0 , sF+F_AB1,0,w01);
    storeC(h1,hT,40,16, sF+F_AB1,0,w01);
    v8f c2={};
    c2=WMMA_F16(ldA(hT,40,0), ldB(wH+AW2,32,0,0), c2);
    storeC(c2,hT,40,0, sF+F_AB2,0,w02);
    v8f c3={};
    c3=WMMA_F16(ldA(hT,40,0), ldB(wH+AW3,32,0,0), c3);
    if (ch<3){
      const float b3=sF[F_AB3+ch];
      #pragma unroll
      for(int r=0;r<8;r++) scr[(r+8*hi)*4+ch]=c3[r]+b3;
    }
    if (l<16){
      float x=scr[l*4],y=scr[l*4+1],z=scr[l*4+2];
      float inv=1.f/fmaxf(sqrtf(x*x+y*y+z*z),1e-12f);
      x*=inv;y*=inv;z*=inv;
      const float s=sF[F_MISC+0];
      x*=s;y*=s;z=z*s+(1.f-s);
      inv=1.f/fmaxf(sqrtf(x*x+y*y+z*z),1e-12f);
      x*=inv;y*=inv;z*=inv;
      const size_t ob=((size_t)b*7)*BHW + hw0 + l;
      out[ob]=x; out[ob+BHW]=y; out[ob+2*(size_t)BHW]=z;
      const float av[3]={x,y,z};
      #pragma unroll
      for(int c=0;c<3;c++){
        enT[l*104+c]=(_Float16)av[c];
        #pragma unroll
        for(int f2=0;f2<4;f2++){
          const float xf=av[c]*(float)(1<<f2);
          enT[l*104+3 +c*4+f2]=(_Float16)__sinf(xf);
          enT[l*104+15+c*4+f2]=(_Float16)__cosf(xf);
        }
      }
    }
  }

  // ---- beta film: normalize(x1), sine; -> normalize -> enc_b ----
  {
    v8f cb = film(wH,sF,0,feT,enT,hT,scr,true,true,0.f);
    if (ch<2){
      #pragma unroll
      for(int r=0;r<8;r++) scr[(r+8*hi)*4+ch]=cb[r];
    }
    if (l<16){
      float x=scr[l*4],y=scr[l*4+1];
      const float inv=1.f/fmaxf(sqrtf(x*x+y*y),1e-12f);
      x*=inv;y*=inv;
      const size_t ob=((size_t)b*7+3)*BHW + hw0 + l;
      out[ob]=x; out[ob+BHW]=y;
      const float av[2]={x,y};
      enT[l*104+27]=(_Float16)x; enT[l*104+28]=(_Float16)y;
      #pragma unroll
      for(int c=0;c<2;c++){
        #pragma unroll
        for(int f2=0;f2<4;f2++){
          const float xf=av[c]*(float)(1<<f2);
          enT[l*104+29+c*4+f2]=(_Float16)__sinf(xf);
          enT[l*104+37+c*4+f2]=(_Float16)__cosf(xf);
        }
      }
    }
  }

  // ---- width film: leaky 0.2; sigmoid to out, raw width -> enc_w (RBF) ----
  {
    v8f cw = film(wH,sF,1,feT,enT,hT,scr,false,false,0.2f);
    if (ch==0){
      #pragma unroll
      for(int r=0;r<8;r++) scr[(r+8*hi)*4]=cw[r];
    }
    if (l<16){
      const float w=scr[l*4];
      out[((size_t)b*7+6)*BHW + hw0 + l]=1.f/(1.f+__expf(-w));
      const float sg=sF[F_MISC+1]; const float i2s=0.5f/(sg*sg);
      #pragma unroll
      for(int c=0;c<10;c++){ float df=w-sF[F_CENT+c];
        enT[l*104+45+c]=(_Float16)__expf(-df*df*i2s); }
    }
  }

  // ---- dist film: leaky 0.2; sigmoid straight to out ----
  {
    v8f cd = film(wH,sF,2,feT,enT,hT,scr,false,false,0.2f);
    if (ch==0){
      const size_t ob=((size_t)b*7+5)*BHW + hw0;
      #pragma unroll
      for(int r=0;r<8;r++){ const int pix=r+8*hi;
        out[ob+pix]=1.f/(1.f+__expf(-cd[r])); }
    }
  }
}

// ---------------- host launch ----------------
// Assumed input order: d_in[0]=features, d_in[1]=depth_, then params leaves in
// sorted-key (jax tree flatten) order:
// 2..9   appr: W1 W2 W3 b1 b2 b3 w0_1 w0_2
// 10..19 beta: W1 W2 W3 Wf b1 b2 b3 bf w0_1 w0_2
// 20     centers
// 21..30 dist: (same leaf order as beta)
// 31 scale, 32 sigma
// 33..42 width: (same leaf order as beta)
extern "C" void kernel_launch(void* const* d_in, const int* in_sizes, int n_in,
                              void* d_out, int out_size, void* d_ws, size_t ws_size,
                              hipStream_t stream){
  (void)in_sizes; (void)n_in; (void)out_size; (void)ws_size;
  auto f32 = [&](int i){ return (const float*)d_in[i]; };
  PackArgs P;
  P.aW1=f32(2); P.aW2=f32(3); P.aW3=f32(4);
  P.aB1=f32(5); P.aB2=f32(6); P.aB3=f32(7); P.aw01=f32(8); P.aw02=f32(9);
  const int idx[3]={10,33,21};                 // beta, width, dist
  for(int f=0;f<3;f++){
    const int o=idx[f];
    P.fW1[f]=f32(o+0); P.fW2[f]=f32(o+1); P.fW3[f]=f32(o+2); P.fWf[f]=f32(o+3);
    P.fB1[f]=f32(o+4); P.fB2[f]=f32(o+5); P.fB3[f]=f32(o+6); P.fBf[f]=f32(o+7);
    P.fw01[f]=f32(o+8); P.fw02[f]=f32(o+9);
  }
  P.centers=f32(20); P.scale=f32(31); P.sigma=f32(32);

  _Float16* wsH=(_Float16*)d_ws;                               // 61440 B
  float*    wsF=(float*)((char*)d_ws + WS_HALVES*sizeof(_Float16)); // +2048 B

  pack_kernel<<<dim3(24,16),256,0,stream>>>(P,wsH,wsF);
  gripper_main<<<GRID,256,0,stream>>>((const float*)d_in[0],(const float*)d_in[1],
                                      wsH,wsF,(float*)d_out);
}
